// MultiHeadAttention_41120016892259
// MI455X (gfx1250) — compile-verified, matrix operand order checked by reading, not run on hardware
//
#include <hip/hip_runtime.h>
#include <stdint.h>

// ---------------------------------------------------------------------------
// Multi-head attention forward for gfx1250 (MI455X), bf16 WMMA pipeline.
// B=2, S=4096, HID=512, HEADS=8, D=64.  Compute-bound -> all three GEMM
// stages through v_wmma_f32_16x16x32_bf16 with f32 accumulation.
// Weight tiles double-buffered in LDS via async global->LDS copies.
// ---------------------------------------------------------------------------

#define BATCH 2
#define SEQ   4096
#define HID   512
#define NHEAD 8
#define HDIM  64
#define BS    (BATCH * SEQ)          // 8192 rows

// gfx1250 async global->LDS path (guarded; falls back to ds_store staging)
#if defined(__has_builtin)
#  if __has_builtin(__builtin_amdgcn_global_load_async_to_lds_b128) && \
      __has_builtin(__builtin_amdgcn_s_wait_asynccnt)
#    define USE_ASYNC_LDS 1
#  endif
#endif
#ifndef USE_ASYNC_LDS
#  define USE_ASYNC_LDS 0
#endif

typedef __attribute__((ext_vector_type(16))) __bf16 v16bf;
typedef __attribute__((ext_vector_type(8)))  float  v8f;

#if USE_ASYNC_LDS
// exact parameter type of the async-load builtin (from clang diagnostic):
// int __attribute__((vector_size(16))) in AS1 (src) / AS3 (dst)
typedef int v4i_vs __attribute__((vector_size(16)));
typedef __attribute__((address_space(1))) v4i_vs* g4ptr;
typedef __attribute__((address_space(3))) v4i_vs* l4ptr;
#endif

union BF16x16 {
    v16bf v;
    uint4 q[2];
};

// round-to-nearest-even f32 -> bf16
static __device__ __forceinline__ __bf16 f2bf(float f) {
    unsigned int u = __builtin_bit_cast(unsigned int, f);
    unsigned int r = (u + 0x7FFFu + ((u >> 16) & 1u)) >> 16;
    unsigned short h = (unsigned short)r;
    return __builtin_bit_cast(__bf16, h);
}

// 16-element bf16 WMMA fragment: 8 contiguous bf16 at p0 (elems 0..7) and 8
// contiguous bf16 at p1 (elems 8..15); 16B-aligned -> *_load_b128 pairs.
static __device__ __forceinline__ v16bf load_frag(const __bf16* p0, const __bf16* p1) {
    BF16x16 u;
    u.q[0] = *(const uint4*)p0;
    u.q[1] = *(const uint4*)p1;
    return u.v;
}

static __device__ __forceinline__ v8f wmma_bf16(v16bf a, v16bf b, v8f c) {
    return __builtin_amdgcn_wmma_f32_16x16x32_bf16(false, a, false, b, (short)0, c,
                                                   false, false);
}

// 16 bytes global -> LDS (async on CDNA5, synchronous staging otherwise)
static __device__ __forceinline__ void copy16_g2l(const __bf16* g, __bf16* l) {
#if USE_ASYNC_LDS
    __builtin_amdgcn_global_load_async_to_lds_b128((g4ptr)(void*)(size_t)(const void*)g,
                                                   (l4ptr)l, 0, 0);
#else
    *(uint4*)l = *(const uint4*)g;
#endif
}

static __device__ __forceinline__ void stage_fence() {
#if USE_ASYNC_LDS
    __builtin_amdgcn_s_wait_asynccnt(0);
#endif
    __syncthreads();
}

// ---------------------------------------------------------------------------
// f32 -> bf16 conversion, 4 elements/thread
// ---------------------------------------------------------------------------
struct alignas(8) BF4 { __bf16 a, b, c, d; };

__global__ void cvt_f32_bf16_kernel(const float* __restrict__ src,
                                    __bf16* __restrict__ dst, int n) {
    int i = (blockIdx.x * blockDim.x + threadIdx.x) * 4;
    if (i + 3 < n) {
        float4 f = *(const float4*)(src + i);
        BF4 o;
        o.a = f2bf(f.x); o.b = f2bf(f.y); o.c = f2bf(f.z); o.d = f2bf(f.w);
        *(BF4*)(dst + i) = o;
    }
}

// ---------------------------------------------------------------------------
// Cooperative stage of a 64(row) x 32(k) bf16 weight tile into LDS.
// tile rows padded to 40 elems (80B) -> conflict-free ds_load_b128 later.
// 128 threads x 2 chunks of 16B = 4KB.
// ---------------------------------------------------------------------------
static __device__ __forceinline__ void stage_wtile(const __bf16* __restrict__ W,
                                                   int n0, int k0,
                                                   __bf16 (*tile)[40], int tid) {
#pragma unroll
    for (int i = 0; i < 2; ++i) {
        const int ch  = tid * 2 + i;       // 0..255
        const int row = ch >> 2;           // 0..63
        const int e   = (ch & 3) * 8;      // 0,8,16,24
        copy16_g2l(W + (size_t)(n0 + row) * HID + k0 + e, &tile[row][e]);
    }
}

// ---------------------------------------------------------------------------
// QKV projection: Y[m,n] = sum_k X[m,k] * W[n,k] + bias[n]
// grid = (BS/128, NHEAD, 3), block = 128 (4 waves). Wave -> 32x64 tile.
// Weight tile double-buffered in LDS, shared by all 4 waves.
// Q,K stored [b,h,s,d]; V stored transposed [b,h,d,t].
// ---------------------------------------------------------------------------
__global__ __launch_bounds__(128)
void qkv_proj_kernel(const __bf16* __restrict__ xb,
                     const __bf16* __restrict__ wqb,
                     const __bf16* __restrict__ wkb,
                     const __bf16* __restrict__ wvb,
                     const float* __restrict__ bq,
                     const float* __restrict__ bk,
                     const float* __restrict__ bv,
                     __bf16* __restrict__ qout,
                     __bf16* __restrict__ kout,
                     __bf16* __restrict__ vtout) {
    __shared__ __bf16 wtile[2][64][40];

    const int tid  = threadIdx.x;
    const int lane = tid & 31;
    const int wave = tid >> 5;
    const int ln   = lane & 15;
    const int off8 = (lane >> 4) ? 8 : 0;

    const int M0    = blockIdx.x * 128 + wave * 32;
    const int head  = blockIdx.y;
    const int which = blockIdx.z;               // 0=Q 1=K 2=V
    const int N0    = head * 64;

    const __bf16* W;
    const float*  bias;
    if (which == 0)      { W = wqb; bias = bq; }
    else if (which == 1) { W = wkb; bias = bk; }
    else                 { W = wvb; bias = bv; }

    stage_wtile(W, N0, 0, wtile[0], tid);
    stage_fence();

    v8f acc[2][4] = {};

    for (int k0 = 0; k0 < HID; k0 += 32) {
        const int buf = (k0 >> 5) & 1;
        if (k0 + 32 < HID)
            stage_wtile(W, N0, k0 + 32, wtile[buf ^ 1], tid);   // prefetch next

        v16bf bfr[4];
#pragma unroll
        for (int nt = 0; nt < 4; ++nt)
            bfr[nt] = load_frag(&wtile[buf][nt * 16 + ln][off8],
                                &wtile[buf][nt * 16 + ln][off8 + 16]);
#pragma unroll
        for (int mt = 0; mt < 2; ++mt) {
            const __bf16* ar = xb + (size_t)(M0 + mt * 16 + ln) * HID + k0 + off8;
            v16bf a = load_frag(ar, ar + 16);
#pragma unroll
            for (int nt = 0; nt < 4; ++nt)
                acc[mt][nt] = wmma_bf16(a, bfr[nt], acc[mt][nt]);
        }
        stage_fence();
    }

    // store: C layout row m = r + (lane<16 ? 0 : 8), col = nt*16 + ln
#pragma unroll
    for (int mt = 0; mt < 2; ++mt) {
#pragma unroll
        for (int nt = 0; nt < 4; ++nt) {
            const int d = nt * 16 + ln;
            const float bval = bias[N0 + d];
#pragma unroll
            for (int r = 0; r < 8; ++r) {
                const int g  = M0 + mt * 16 + r + off8;  // global row (b*S + s)
                const int bi = g >> 12;
                const int si = g & (SEQ - 1);
                const int bh = bi * NHEAD + head;
                const float v = acc[mt][nt][r] + bval;
                if (which == 2) {
                    vtout[((size_t)bh * HDIM + d) * SEQ + si] = f2bf(v);
                } else {
                    __bf16* dst = (which == 0) ? qout : kout;
                    dst[((size_t)bh * SEQ + si) * HDIM + d] = f2bf(v);
                }
            }
        }
    }
}

// ---------------------------------------------------------------------------
// Flash attention: grid = (SEQ/128, NHEAD, BATCH), block = 128 (4 waves).
// Each wave: 32 query rows (2 M-tiles), online softmax over 4096 keys in
// 32-key blocks.  16 WMMAs per key block.
// ---------------------------------------------------------------------------
__global__ __launch_bounds__(128)
void flash_attn_kernel(const __bf16* __restrict__ qb,
                       const __bf16* __restrict__ kb,
                       const __bf16* __restrict__ vT,
                       __bf16* __restrict__ ctxb) {
    __shared__ __bf16 lds_p[4][32][40];          // per-wave padded P tile

    const int lane = threadIdx.x & 31;
    const int wave = threadIdx.x >> 5;
    const int ln   = lane & 15;
    const int off8 = (lane >> 4) ? 8 : 0;

    const int b    = blockIdx.z;
    const int head = blockIdx.y;
    const int bh   = b * NHEAD + head;
    const int q0   = blockIdx.x * 128 + wave * 32;

    const __bf16* qbase = qb + (size_t)bh * SEQ * HDIM;
    const __bf16* kbase = kb + (size_t)bh * SEQ * HDIM;
    const __bf16* vbase = vT + (size_t)bh * HDIM * SEQ;

    // Q A-fragments held in registers for the whole key loop
    v16bf aq[2][2];
#pragma unroll
    for (int mt = 0; mt < 2; ++mt) {
        const __bf16* qrow = qbase + (size_t)(q0 + mt * 16 + ln) * HDIM;
        aq[mt][0] = load_frag(qrow + off8,      qrow + off8 + 16);
        aq[mt][1] = load_frag(qrow + 32 + off8, qrow + 32 + off8 + 16);
    }

    float mrow[2][8], lrow[2][8];
#pragma unroll
    for (int mt = 0; mt < 2; ++mt)
#pragma unroll
        for (int r = 0; r < 8; ++r) { mrow[mt][r] = -3.0e38f; lrow[mt][r] = 0.0f; }
    v8f o[2][4] = {};

    const float scale = 0.125f;                  // 1/sqrt(64)

    for (int t0 = 0; t0 < SEQ; t0 += 32) {
        if (t0 + 32 < SEQ) {                     // warm next K tile (L2->WGP$)
            __builtin_prefetch(kbase + (size_t)(t0 + 32 + ln) * HDIM, 0, 1);
            __builtin_prefetch(kbase + (size_t)(t0 + 48 + ln) * HDIM, 0, 1);
        }

        // ---- scores: S = Q * K^T  (4 C tiles, K-dim 64 = 2 wmmas each)
        v16bf bk[2][2];
#pragma unroll
        for (int tt = 0; tt < 2; ++tt) {
            const __bf16* kr = kbase + (size_t)(t0 + tt * 16 + ln) * HDIM;
            bk[tt][0] = load_frag(kr + off8,      kr + off8 + 16);
            bk[tt][1] = load_frag(kr + 32 + off8, kr + 32 + off8 + 16);
        }
        v8f s[2][2] = {};
#pragma unroll
        for (int mt = 0; mt < 2; ++mt)
#pragma unroll
            for (int tt = 0; tt < 2; ++tt) {
                s[mt][tt] = wmma_bf16(aq[mt][0], bk[tt][0], s[mt][tt]);
                s[mt][tt] = wmma_bf16(aq[mt][1], bk[tt][1], s[mt][tt]);
            }

        // ---- online softmax + accumulator rescale + P staging
#pragma unroll
        for (int mt = 0; mt < 2; ++mt) {
#pragma unroll
            for (int r = 0; r < 8; ++r) {
                const float x0 = s[mt][0][r] * scale;
                const float x1 = s[mt][1][r] * scale;
                float mx = fmaxf(x0, x1);
#pragma unroll
                for (int sh = 1; sh < 16; sh <<= 1)
                    mx = fmaxf(mx, __shfl_xor(mx, sh, 32));
                const float mnew  = fmaxf(mrow[mt][r], mx);
                const float alpha = __expf(mrow[mt][r] - mnew);
                mrow[mt][r] = mnew;
                const float p0 = __expf(x0 - mnew);
                const float p1 = __expf(x1 - mnew);
                float ps = p0 + p1;
#pragma unroll
                for (int sh = 1; sh < 16; sh <<= 1)
                    ps += __shfl_xor(ps, sh, 32);
                lrow[mt][r] = lrow[mt][r] * alpha + ps;
#pragma unroll
                for (int nt = 0; nt < 4; ++nt)
                    o[mt][nt][r] *= alpha;
                const int m = mt * 16 + r + off8;
                lds_p[wave][m][ln]      = f2bf(p0);
                lds_p[wave][m][16 + ln] = f2bf(p1);
            }
        }
        __syncthreads();

        // ---- read P back in A-fragment layout
        v16bf ap[2];
#pragma unroll
        for (int mt = 0; mt < 2; ++mt)
            ap[mt] = load_frag(&lds_p[wave][mt * 16 + ln][off8],
                               &lds_p[wave][mt * 16 + ln][off8 + 16]);
        __syncthreads();

        // ---- O += P * V  (V is [d, t] -> contiguous B loads, shared by M-tiles)
#pragma unroll
        for (int nt = 0; nt < 4; ++nt) {
            const __bf16* vr = vbase + (size_t)(nt * 16 + ln) * SEQ + t0 + off8;
            v16bf bv = load_frag(vr, vr + 16);
            o[0][nt] = wmma_bf16(ap[0], bv, o[0][nt]);
            o[1][nt] = wmma_bf16(ap[1], bv, o[1][nt]);
        }
    }

    // ---- epilogue: ctx[b, s, head*64 + d] = o / l
#pragma unroll
    for (int mt = 0; mt < 2; ++mt)
#pragma unroll
        for (int r = 0; r < 8; ++r) {
            const int srow = q0 + mt * 16 + r + off8;
            const float inv = 1.0f / lrow[mt][r];
#pragma unroll
            for (int nt = 0; nt < 4; ++nt) {
                const float v = o[mt][nt][r] * inv;
                ctxb[((size_t)(b * SEQ + srow)) * HID + head * 64 + nt * 16 + ln] =
                    f2bf(v);
            }
        }
}

// ---------------------------------------------------------------------------
// Output projection: out[m,n] = sum_k ctx[m,k] * Wo[n,k] + bo[n]   (f32 out)
// grid = (BS/128, HID/64), block = 128.
// ---------------------------------------------------------------------------
__global__ __launch_bounds__(128)
void out_proj_kernel(const __bf16* __restrict__ ctxb,
                     const __bf16* __restrict__ wob,
                     const float* __restrict__ bo,
                     float* __restrict__ out) {
    __shared__ __bf16 wtile[2][64][40];

    const int tid  = threadIdx.x;
    const int lane = tid & 31;
    const int wave = tid >> 5;
    const int ln   = lane & 15;
    const int off8 = (lane >> 4) ? 8 : 0;

    const int M0 = blockIdx.x * 128 + wave * 32;
    const int N0 = blockIdx.y * 64;

    stage_wtile(wob, N0, 0, wtile[0], tid);
    stage_fence();

    v8f acc[2][4] = {};

    for (int k0 = 0; k0 < HID; k0 += 32) {
        const int buf = (k0 >> 5) & 1;
        if (k0 + 32 < HID)
            stage_wtile(wob, N0, k0 + 32, wtile[buf ^ 1], tid);

        v16bf bfr[4];
#pragma unroll
        for (int nt = 0; nt < 4; ++nt)
            bfr[nt] = load_frag(&wtile[buf][nt * 16 + ln][off8],
                                &wtile[buf][nt * 16 + ln][off8 + 16]);
#pragma unroll
        for (int mt = 0; mt < 2; ++mt) {
            const __bf16* ar = ctxb + (size_t)(M0 + mt * 16 + ln) * HID + k0 + off8;
            v16bf a = load_frag(ar, ar + 16);
#pragma unroll
            for (int nt = 0; nt < 4; ++nt)
                acc[mt][nt] = wmma_bf16(a, bfr[nt], acc[mt][nt]);
        }
        stage_fence();
    }

#pragma unroll
    for (int mt = 0; mt < 2; ++mt)
#pragma unroll
        for (int nt = 0; nt < 4; ++nt) {
            const int n = N0 + nt * 16 + ln;
            const float bval = bo[n];
#pragma unroll
            for (int r = 0; r < 8; ++r) {
                const int m = M0 + mt * 16 + r + off8;
                out[(size_t)m * HID + n] = acc[mt][nt][r] + bval;
            }
        }
}

// ---------------------------------------------------------------------------
// Host launcher
// ---------------------------------------------------------------------------
extern "C" void kernel_launch(void* const* d_in, const int* in_sizes, int n_in,
                              void* d_out, int out_size, void* d_ws, size_t ws_size,
                              hipStream_t stream) {
    (void)in_sizes; (void)n_in; (void)out_size; (void)ws_size;

    const float* x  = (const float*)d_in[0];
    const float* Wq = (const float*)d_in[1];
    const float* bq = (const float*)d_in[2];
    const float* Wk = (const float*)d_in[3];
    const float* bk = (const float*)d_in[4];
    const float* Wv = (const float*)d_in[5];
    const float* bv = (const float*)d_in[6];
    const float* Wo = (const float*)d_in[7];
    const float* bo = (const float*)d_in[8];
    float* out = (float*)d_out;

    // workspace layout (bf16 scratch, ~42 MB total)
    const size_t XB_ELE = (size_t)BS * HID;       // 4,194,304
    const size_t W_ELE  = (size_t)HID * HID;      //   262,144
    char* ws = (char*)d_ws;
    size_t off = 0;
    __bf16* xb   = (__bf16*)(ws + off); off += XB_ELE * 2;
    __bf16* wqb  = (__bf16*)(ws + off); off += W_ELE * 2;
    __bf16* wkb  = (__bf16*)(ws + off); off += W_ELE * 2;
    __bf16* wvb  = (__bf16*)(ws + off); off += W_ELE * 2;
    __bf16* wob  = (__bf16*)(ws + off); off += W_ELE * 2;
    __bf16* qbuf = (__bf16*)(ws + off); off += XB_ELE * 2;
    __bf16* kbuf = (__bf16*)(ws + off); off += XB_ELE * 2;
    __bf16* vtb  = (__bf16*)(ws + off); off += XB_ELE * 2;
    __bf16* ctxb = (__bf16*)(ws + off); off += XB_ELE * 2;

    // 1) f32 -> bf16 conversions
    const int CT = 256;
    cvt_f32_bf16_kernel<<<(int)(XB_ELE / 4 + CT - 1) / CT, CT, 0, stream>>>(x,  xb,  (int)XB_ELE);
    cvt_f32_bf16_kernel<<<(int)(W_ELE  / 4 + CT - 1) / CT, CT, 0, stream>>>(Wq, wqb, (int)W_ELE);
    cvt_f32_bf16_kernel<<<(int)(W_ELE  / 4 + CT - 1) / CT, CT, 0, stream>>>(Wk, wkb, (int)W_ELE);
    cvt_f32_bf16_kernel<<<(int)(W_ELE  / 4 + CT - 1) / CT, CT, 0, stream>>>(Wv, wvb, (int)W_ELE);
    cvt_f32_bf16_kernel<<<(int)(W_ELE  / 4 + CT - 1) / CT, CT, 0, stream>>>(Wo, wob, (int)W_ELE);

    // 2) fused QKV projection (Q,K row-major per head; V transposed per head)
    dim3 gqkv(BS / 128, NHEAD, 3);
    qkv_proj_kernel<<<gqkv, 128, 0, stream>>>(xb, wqb, wkb, wvb, bq, bk, bv,
                                              qbuf, kbuf, vtb);

    // 3) flash attention
    dim3 gattn(SEQ / 128, NHEAD, BATCH);
    flash_attn_kernel<<<gattn, 128, 0, stream>>>(qbuf, kbuf, vtb, ctxb);

    // 4) output projection -> f32 out
    dim3 gout(BS / 128, HID / 64);
    out_proj_kernel<<<gout, 128, 0, stream>>>(ctxb, wob, bo, out);
}
